// pLayer_89335319757268
// MI455X (gfx1250) — matrix operand order repaired and checked by reading, not compiled, not run on hardware
//
#include <hip/hip_runtime.h>
#include <math.h>

typedef float  v8f   __attribute__((ext_vector_type(8)));
typedef __bf16 v16bf __attribute__((ext_vector_type(16)));
typedef __bf16 v8bf  __attribute__((ext_vector_type(8)));

#define N_IN    1024
#define N_OUT   1024
#define EBATCH  8192
#define MROWS   1026
#define MP      1056          // MROWS padded to multiple of 32
#define KTOT    (2*MP)        // 2112
#define GMIN_C  0.01f
#define GMAX_C  10.0f
#define PGMIN_C 0.1f
#define EPS_C   1e-10f

#define BM 128
#define BN 128
#define BK 32
#define LDA 40                // LDS row padding (bf16 elems) to kill bank conflicts

__device__ __forceinline__ float tanhc(float x, float e0, float e1, float e2, float e3) {
    return e0 + e1 * tanhf((x - e2) * e3);
}

// ---------------- K1: per-column theta stats: denom, min|theta_|, gcol ----------------
__global__ void k_col_stats(const float* __restrict__ th, float* denom, float* minv, float* gcol) {
    int n = blockIdx.x * blockDim.x + threadIdx.x;   // 0..1023
    float sraw = 0.f, sclip = 0.f, mn = 3.4e38f;
    for (int m = 0; m < MROWS; ++m) {
        float v  = th[(size_t)m * N_OUT + n];
        float av = fabsf(v);
        sraw += av;
        mn = fminf(mn, av);
        float t = fminf(fmaxf(v, -GMAX_C), GMAX_C);
        if (fabsf(t) < GMIN_C) t = 0.f;
        sclip += fabsf(t);
    }
    denom[n] = sclip + EPS_C;
    minv[n]  = mn;
    gcol[n]  = PGMIN_C * sraw / mn;   // sum_m g_tilde[m,n]
}

// ---------------- K2: per-row sums of Gp, Gn ----------------
__global__ void k_row_gsums(const float* __restrict__ th, const float* __restrict__ minv,
                            float* gp, float* gn) {
    int m = blockIdx.x;               // 0..1025
    float p = 0.f, q = 0.f;
    for (int n = threadIdx.x; n < N_OUT; n += 256) {
        float v = th[(size_t)m * N_OUT + n];
        float g = fabsf(v) * (PGMIN_C / minv[n]);
        float t = fminf(fmaxf(v, -GMAX_C), GMAX_C);
        if (fabsf(t) < GMIN_C) t = 0.f;
        if (t >= 0.f) p += g; else q += g;
    }
    __shared__ float rp[256], rq[256];
    rp[threadIdx.x] = p; rq[threadIdx.x] = q;
    __syncthreads();
    for (int s = 128; s > 0; s >>= 1) {
        if (threadIdx.x < s) { rp[threadIdx.x] += rp[threadIdx.x + s]; rq[threadIdx.x] += rq[threadIdx.x + s]; }
        __syncthreads();
    }
    if (threadIdx.x == 0) { gp[m] = rp[0]; gn[m] = rq[0]; }
}

// ---------------- K3: build B operands [N][KTOT] bf16 hi/lo (k-contiguous) ----------------
__global__ void k_build_B(const float* __restrict__ th, const float* __restrict__ denom,
                          const float* __restrict__ minv,
                          __bf16* __restrict__ Wh, __bf16* __restrict__ Wl,
                          __bf16* __restrict__ Gh, __bf16* __restrict__ Gl) {
    __shared__ float Wf[32][33], Gf[32][33];
    int m0 = blockIdx.x * 32, n0 = blockIdx.y * 32;
    for (int l = threadIdx.x; l < 1024; l += 256) {
        int mi = l >> 5, ni = l & 31;               // coalesced read over n
        int m = m0 + mi, n = n0 + ni;
        float v = (m < MROWS) ? th[(size_t)m * N_OUT + n] : 0.f;
        float t = fminf(fmaxf(v, -GMAX_C), GMAX_C);
        if (fabsf(t) < GMIN_C) t = 0.f;
        bool pos = (t >= 0.f);
        float w = fabsf(t) / denom[n];
        float g = fabsf(v) * (PGMIN_C / minv[n]);
        Wf[mi][ni] = pos ? w : -w;                  // sign encodes pos/neg
        Gf[mi][ni] = pos ? g : -g;
    }
    __syncthreads();
    for (int l = threadIdx.x; l < 1024; l += 256) {
        int ni = l >> 5, mi = l & 31;               // coalesced-ish write over k
        int n = n0 + ni, k = m0 + mi;
        float wsv = Wf[mi][ni], gsv = Gf[mi][ni];
        float wp, wn, gpv, gnv;
        if (wsv >= 0.f) { wp = wsv; wn = 0.f; } else { wp = 0.f; wn = -wsv; }
        if (gsv >= 0.f) { gpv = gsv; gnv = 0.f; } else { gpv = 0.f; gnv = -gsv; }
        size_t base = (size_t)n * KTOT;
        __bf16 h;
        h = (__bf16)wp;  Wh[base + k]      = h; Wl[base + k]      = (__bf16)(wp  - (float)h);
        h = (__bf16)wn;  Wh[base + MP + k] = h; Wl[base + MP + k] = (__bf16)(wn  - (float)h);
        h = (__bf16)gpv; Gh[base + k]      = h; Gl[base + k]      = (__bf16)(gpv - (float)h);
        h = (__bf16)gnv; Gh[base + MP + k] = h; Gl[base + MP + k] = (__bf16)(gnv - (float)h);
    }
}

// ---------------- K4: partial column sums of x_ext^2 and x_neg^2 ----------------
__global__ void k_xstats_part(const float* __restrict__ a, const float* __restrict__ eta_inv,
                              float* __restrict__ ppart, float* __restrict__ qpart) {
    int c = blockIdx.x * 256 + threadIdx.x;   // column 0..1023
    int chunk = blockIdx.y;                   // 0..31
    float ei0 = eta_inv[0], ei1 = eta_inv[1], ei2 = eta_inv[2], ei3 = eta_inv[3];
    float p = 0.f, q = 0.f;
    int e0 = chunk * 256;
    for (int e = e0; e < e0 + 256; ++e) {
        float v = a[(size_t)e * N_IN + c];
        p += v * v;
        float xn = tanhc(v, ei0, ei1, ei2, ei3);
        q += xn * xn;
    }
    ppart[(size_t)chunk * N_IN + c] = p;
    qpart[(size_t)chunk * N_IN + c] = q;
}

// ---------------- K4b: reduce partials + special rows (ones/zeros cols) ----------------
__global__ void k_xstats_reduce(const float* __restrict__ ppart, const float* __restrict__ qpart,
                                const float* __restrict__ eta_inv,
                                float* __restrict__ sx2p, float* __restrict__ sx2n) {
    int c = blockIdx.x * 256 + threadIdx.x;
    float p = 0.f, q = 0.f;
    for (int ch = 0; ch < 32; ++ch) { p += ppart[(size_t)ch * N_IN + c]; q += qpart[(size_t)ch * N_IN + c]; }
    sx2p[c] = p; sx2n[c] = q;
    if (blockIdx.x == 0 && threadIdx.x < 32) {
        int m = 1024 + threadIdx.x;
        float vp = 0.f, vq = 0.f;
        if (threadIdx.x == 0) {   // m = 1024 (ones column)
            vp = (float)EBATCH;
            float xn1 = tanhc(1.f, eta_inv[0], eta_inv[1], eta_inv[2], eta_inv[3]);
            vq = (float)EBATCH * xn1 * xn1;
        }
        sx2p[m] = vp; sx2n[m] = vq;   // m=1025 (zeros col, x_neg forced 0) and padding -> 0
    }
}

// ---------------- K5: fused dual-GEMM (z, u) + activation + partial reductions ----------------
__global__ __launch_bounds__(256) void k_gemm_fused(
    const float* __restrict__ a,
    const __bf16* __restrict__ Wh, const __bf16* __restrict__ Wl,
    const __bf16* __restrict__ Gh, const __bf16* __restrict__ Gl,
    const float* __restrict__ gcol,
    const float* __restrict__ eta_act, const float* __restrict__ eta_inv,
    float* __restrict__ out, float* __restrict__ s3_part, float* __restrict__ cross_part)
{
    __shared__ __align__(16) __bf16 Ah[BM][LDA];
    __shared__ __align__(16) __bf16 Al[BM][LDA];
    __shared__ float red[256];

    const int tid  = threadIdx.x;
    const int lane = tid & 31;
    const int wid  = tid >> 5;                 // 0..7 -> wave's 16-col slice
    const int eBase = blockIdx.x * BM;
    const int nBase = blockIdx.y * BN;

    const float ei0 = eta_inv[0], ei1 = eta_inv[1], ei2 = eta_inv[2], ei3 = eta_inv[3];
    const float xn1 = tanhc(1.f, ei0, ei1, ei2, ei3);

    v8f accZ[8], accU[8];
#pragma unroll
    for (int j = 0; j < 8; ++j) { accZ[j] = 0.f; accU[j] = 0.f; }

    // B fragment addressing: lane -> column nB, contiguous K-half per lane (k = 16*(lane/16) + elem)
    const int nB = nBase + wid * 16 + (lane & 15);
    const int kHalf = (lane >> 4) << 4;        // lanes 0-15: K+0..15, lanes 16-31: K+16..31
    const size_t rowB = (size_t)nB * KTOT;

    // A staging: 2 threads per row, 16 elems each
    const int sr = tid >> 1;
    const int sk = (tid & 1) * 16;

    // A fragment addressing: row lane%16, contiguous k = 16*(lane/16) + elem (same as B)
    const int mA = lane & 15;
    const int kA = (lane < 16) ? 0 : 16;

    for (int s = 0; s < KTOT / BK; ++s) {
        __syncthreads();
        {   // stage A tile (x_ext for k<MP, x_neg for k>=MP), split into bf16 hi/lo
            const int e = eBase + sr;
            const int kg0 = s * BK + sk;
            if (s <= 31) {                       // pure a loads (x_ext)
                const float* ap = a + (size_t)e * N_IN + kg0;
#pragma unroll
                for (int i = 0; i < 16; ++i) {
                    float x = ap[i];
                    __bf16 h = (__bf16)x;
                    Ah[sr][sk + i] = h; Al[sr][sk + i] = (__bf16)(x - (float)h);
                }
            } else if (s == 32) {                // x_ext constants: ones col + zero pad
#pragma unroll
                for (int i = 0; i < 16; ++i) {
                    float x = ((kg0 + i) == 1024) ? 1.f : 0.f;
                    __bf16 h = (__bf16)x;
                    Ah[sr][sk + i] = h; Al[sr][sk + i] = (__bf16)(x - (float)h);
                }
            } else if (s <= 64) {                // x_neg = tanh circuit of a
                const int m0 = kg0 - MP;
                const float* ap = a + (size_t)e * N_IN + m0;
#pragma unroll
                for (int i = 0; i < 16; ++i) {
                    float x = tanhc(ap[i], ei0, ei1, ei2, ei3);
                    __bf16 h = (__bf16)x;
                    Ah[sr][sk + i] = h; Al[sr][sk + i] = (__bf16)(x - (float)h);
                }
            } else {                             // s==65: x_neg constants (col 1025 forced 0)
                const int m0 = kg0 - MP;
#pragma unroll
                for (int i = 0; i < 16; ++i) {
                    float x = ((m0 + i) == 1024) ? xn1 : 0.f;
                    __bf16 h = (__bf16)x;
                    Ah[sr][sk + i] = h; Al[sr][sk + i] = (__bf16)(x - (float)h);
                }
            }
        }
        __syncthreads();

        const size_t ob = rowB + (size_t)s * BK + kHalf;  // 32B-aligned contiguous per lane
        v16bf bWh = *(const v16bf*)(Wh + ob);
        v16bf bWl = *(const v16bf*)(Wl + ob);
        v16bf bGh = *(const v16bf*)(Gh + ob);
        v16bf bGl = *(const v16bf*)(Gl + ob);

#pragma unroll
        for (int j = 0; j < 8; ++j) {
            const int r = j * 16 + mA;
            v8bf h0 = *(const v8bf*)(&Ah[r][kA]);
            v8bf h1 = *(const v8bf*)(&Ah[r][kA + 8]);
            v8bf l0 = *(const v8bf*)(&Al[r][kA]);
            v8bf l1 = *(const v8bf*)(&Al[r][kA + 8]);
            v16bf aH = __builtin_shufflevector(h0, h1, 0,1,2,3,4,5,6,7,8,9,10,11,12,13,14,15);
            v16bf aL = __builtin_shufflevector(l0, l1, 0,1,2,3,4,5,6,7,8,9,10,11,12,13,14,15);
            // z += x * Wcat  (bf16x3 split for ~fp32 accuracy)
            accZ[j] = __builtin_amdgcn_wmma_f32_16x16x32_bf16(false, aH, false, bWh, (short)0, accZ[j], false, false);
            accZ[j] = __builtin_amdgcn_wmma_f32_16x16x32_bf16(false, aH, false, bWl, (short)0, accZ[j], false, false);
            accZ[j] = __builtin_amdgcn_wmma_f32_16x16x32_bf16(false, aL, false, bWh, (short)0, accZ[j], false, false);
            // u += x * Gcat
            accU[j] = __builtin_amdgcn_wmma_f32_16x16x32_bf16(false, aH, false, bGh, (short)0, accU[j], false, false);
            accU[j] = __builtin_amdgcn_wmma_f32_16x16x32_bf16(false, aH, false, bGl, (short)0, accU[j], false, false);
            accU[j] = __builtin_amdgcn_wmma_f32_16x16x32_bf16(false, aL, false, bGh, (short)0, accU[j], false, false);
        }
    }

    // epilogue: a_new = tanh_circuit(z, eta_act); partials for S3 and cross
    const float ea0 = eta_act[0], ea1 = eta_act[1], ea2 = eta_act[2], ea3 = eta_act[3];
    const float gc = gcol[nB];
    const int eOff = eBase + ((lane < 16) ? 0 : 8);
    float s3 = 0.f, cr = 0.f;
#pragma unroll
    for (int j = 0; j < 8; ++j) {
#pragma unroll
        for (int r = 0; r < 8; ++r) {
            float z = accZ[j][r], u = accU[j][r];
            int e = eOff + j * 16 + r;
            out[(size_t)e * N_OUT + nB] = ea0 + ea1 * tanhf((z - ea2) * ea3);
            s3 += gc * z * z;
            cr += u * z;
        }
    }
    red[tid] = s3; __syncthreads();
    for (int st = 128; st > 0; st >>= 1) { if (tid < st) red[tid] += red[tid + st]; __syncthreads(); }
    float s3tot = red[0];
    __syncthreads();
    red[tid] = cr; __syncthreads();
    for (int st = 128; st > 0; st >>= 1) { if (tid < st) red[tid] += red[tid + st]; __syncthreads(); }
    if (tid == 0) {
        int b = blockIdx.y * gridDim.x + blockIdx.x;
        s3_part[b] = s3tot;
        cross_part[b] = red[0];
    }
}

// ---------------- K6: final scalar assembly ----------------
__global__ void k_final(const float* gp, const float* gn, const float* sx2p, const float* sx2n,
                        const float* s3_part, const float* cross_part, int nblk,
                        float* out_scalar) {
    float acc = 0.f;
    for (int m = threadIdx.x; m < MROWS; m += 256) acc += gp[m] * sx2p[m] + gn[m] * sx2n[m];
    for (int b = threadIdx.x; b < nblk; b += 256) acc += s3_part[b] - 2.f * cross_part[b];
    __shared__ float red[256];
    red[threadIdx.x] = acc; __syncthreads();
    for (int s = 128; s > 0; s >>= 1) { if (threadIdx.x < s) red[threadIdx.x] += red[threadIdx.x + s]; __syncthreads(); }
    if (threadIdx.x == 0) *out_scalar = red[0] / (float)EBATCH;
}

extern "C" void kernel_launch(void* const* d_in, const int* in_sizes, int n_in,
                              void* d_out, int out_size, void* d_ws, size_t ws_size,
                              hipStream_t stream) {
    (void)in_sizes; (void)n_in; (void)out_size; (void)ws_size;
    const float* a       = (const float*)d_in[0];
    const float* theta   = (const float*)d_in[1];
    const float* eta_act = (const float*)d_in[2];
    const float* eta_inv = (const float*)d_in[3];
    float* out = (float*)d_out;

    char* w = (char*)d_ws;
    size_t off = 0;
    auto balloc = [&](size_t bytes) -> char* {
        char* p = w + off;
        off = (off + bytes + 255) & ~(size_t)255;
        return p;
    };
    float* denom = (float*)balloc(1024 * sizeof(float));
    float* minv  = (float*)balloc(1024 * sizeof(float));
    float* gcol  = (float*)balloc(1024 * sizeof(float));
    float* gp    = (float*)balloc(MP * sizeof(float));
    float* gn    = (float*)balloc(MP * sizeof(float));
    float* sx2p  = (float*)balloc(MP * sizeof(float));
    float* sx2n  = (float*)balloc(MP * sizeof(float));
    float* ppart = (float*)balloc((size_t)32 * 1024 * sizeof(float));
    float* qpart = (float*)balloc((size_t)32 * 1024 * sizeof(float));
    const int NBLK = (EBATCH / BM) * (N_OUT / BN);   // 512
    float* s3p = (float*)balloc(NBLK * sizeof(float));
    float* crp = (float*)balloc(NBLK * sizeof(float));
    __bf16* Wh = (__bf16*)balloc((size_t)N_OUT * KTOT * 2);
    __bf16* Wl = (__bf16*)balloc((size_t)N_OUT * KTOT * 2);
    __bf16* Gh = (__bf16*)balloc((size_t)N_OUT * KTOT * 2);
    __bf16* Gl = (__bf16*)balloc((size_t)N_OUT * KTOT * 2);

    k_col_stats<<<dim3(4), dim3(256), 0, stream>>>(theta, denom, minv, gcol);
    k_row_gsums<<<dim3(MROWS), dim3(256), 0, stream>>>(theta, minv, gp, gn);
    k_build_B<<<dim3(MP / 32, 32), dim3(256), 0, stream>>>(theta, denom, minv, Wh, Wl, Gh, Gl);
    k_xstats_part<<<dim3(4, 32), dim3(256), 0, stream>>>(a, eta_inv, ppart, qpart);
    k_xstats_reduce<<<dim3(4), dim3(256), 0, stream>>>(ppart, qpart, eta_inv, sx2p, sx2n);
    k_gemm_fused<<<dim3(EBATCH / BM, N_OUT / BN), dim3(256), 0, stream>>>(
        a, Wh, Wl, Gh, Gl, gcol, eta_act, eta_inv, out, s3p, crp);
    k_final<<<dim3(1), dim3(256), 0, stream>>>(gp, gn, sx2p, sx2n, s3p, crp, NBLK,
                                               out + (size_t)EBATCH * N_OUT);
}
